// MSEConnLossMultiCube_22771916603647
// MI455X (gfx1250) — compile-verified, matrix operand order checked by reading, not run on hardware
//
#include <hip/hip_runtime.h>

typedef __attribute__((ext_vector_type(16))) _Float16 v16h;
typedef __attribute__((ext_vector_type(8)))  float    v8f;
typedef __attribute__((ext_vector_type(4)))  unsigned int v4u;
typedef __attribute__((ext_vector_type(8)))  int      v8i;
typedef __attribute__((ext_vector_type(4)))  int      v4i;

#define NB_MSE   1024
#define NTASK    384
#define ALPHA_C  0.0001f
#define BETA_C   0.1f
#define DMAX_C   20.0f

// ---------------------------------------------------------------------------
// Block-wide sum for 256-thread blocks (8 wave32 waves). Result broadcast.
// ---------------------------------------------------------------------------
__device__ __forceinline__ float block_sum(float v, float* red) {
  __syncthreads();                       // protect red from previous use
  int lane = threadIdx.x & 31;
  int wid  = threadIdx.x >> 5;
  #pragma unroll
  for (int off = 16; off > 0; off >>= 1) v += __shfl_down(v, off, 32);
  if (lane == 0) red[wid] = v;
  __syncthreads();
  if (threadIdx.x == 0) {
    float t = 0.f;
    int nw = blockDim.x >> 5;
    for (int w = 0; w < nw; ++w) t += red[w];
    red[0] = t;
  }
  __syncthreads();
  return red[0];
}

// ---------------------------------------------------------------------------
// TDM 2D tile load: global (row-major, line length 64 f32) -> LDS, 32x32 tile.
// Issued once per calling wave; caller must gate to a single wave and then
// s_wait_tensorcnt + barrier before consumers read LDS.
// This toolchain exposes the 6-arg builtin:
//   (uint32x4 g0, int32x8 g1, int32x4 g2, int32x4 g3, int32x8 pad, i32 cpol)
// ---------------------------------------------------------------------------
__device__ __forceinline__ void tdm_load_tile_32x32_f32(unsigned lds_off,
                                                        unsigned long long ga) {
#if __has_builtin(__builtin_amdgcn_tensor_load_to_lds)
  v4u g0 = { 1u,                                    // count=1, user mode
             lds_off,                               // lds_addr (bytes)
             (unsigned)(ga & 0xFFFFFFFFull),        // global_addr[31:0]
             (unsigned)((ga >> 32) & 0x1FFFFFFull)  // global_addr[56:32]
               | 0x80000000u };                     // type=2 ("image")
  v8i g1 = { 0x00020000,          // data_size=4B, mask=0, no pad/iter/barrier
             (int)(64u << 16),    // tensor_dim0 = 64 (bits 63:48)
             (int)(64u << 16),    // tensor_dim1 = 64 (bits 111:80 low half)
             (int)(32u << 16),    // tile_dim0   = 32 (bits 127:112)
             32,                  // tile_dim1   = 32 (bits 143:128)
             64,                  // tensor_dim0_stride = 64 (bits 191:160)
             (int)(4096u << 16),  // tensor_dim1_stride = 4096 (low 16)
             0 };
  v4i gz4 = { 0, 0, 0, 0 };       // groups 2/3 unused for 2D tiles
  v8i gz8 = { 0, 0, 0, 0, 0, 0, 0, 0 };
  __builtin_amdgcn_tensor_load_to_lds(g0, g1, gz4, gz4, gz8, 0);
#else
  (void)lds_off; (void)ga;
#endif
}

// ---------------------------------------------------------------------------
// Kernel 1: MSE partial sums via WMMA ones-contraction.
// Guard-free full-tile loop (wave-uniform bounds): 512 elements/wave/step.
// ---------------------------------------------------------------------------
__global__ __launch_bounds__(256) void k_mse(const float* __restrict__ p,
                                             const float* __restrict__ t,
                                             float* __restrict__ partials,
                                             long long N) {
  __shared__ float red[8];
  v16h ones;
  #pragma unroll
  for (int i = 0; i < 16; ++i) ones[i] = (_Float16)1.0f;
  v8f c = {};

  int lane = threadIdx.x & 31;
  long long gw     = (((long long)blockIdx.x * blockDim.x) + threadIdx.x) >> 5;
  long long nWaves = ((long long)gridDim.x * blockDim.x) >> 5;
  long long stride = nWaves * 512;
  long long nFull  = (N / 512) * 512;       // N = 8,388,608 -> nFull == N

  for (long long w0 = gw * 512; w0 < nFull; w0 += stride) {  // uniform bound
    long long pos = w0 + (long long)lane * 16;
    v16h a;
    #pragma unroll
    for (int q = 0; q < 4; ++q) {
      long long ix = pos + q * 4;
      float4 pv = *reinterpret_cast<const float4*>(p + ix);
      float4 tv = *reinterpret_cast<const float4*>(t + ix);
      float d0 = pv.x - tv.x, d1 = pv.y - tv.y;
      float d2 = pv.z - tv.z, d3 = pv.w - tv.w;
      a[q*4+0] = (_Float16)(d0*d0); a[q*4+1] = (_Float16)(d1*d1);
      a[q*4+2] = (_Float16)(d2*d2); a[q*4+3] = (_Float16)(d3*d3);
    }
    if (w0 + stride < nFull) {
      __builtin_prefetch(p + pos + stride, 0, 1);
      __builtin_prefetch(t + pos + stride, 0, 1);
    }
    c = __builtin_amdgcn_wmma_f32_16x16x32_f16(false, a, false, ones,
                                               (short)0, c, false, false);
  }

  // ragged tail (empty when N % 512 == 0), exact f32
  float extra = 0.f;
  long long gtid = (long long)blockIdx.x * blockDim.x + threadIdx.x;
  long long nthr = (long long)gridDim.x * blockDim.x;
  for (long long i = nFull + gtid; i < N; i += nthr) {
    float d = p[i] - t[i];
    extra += d * d;
  }

  float s = (c[0]+c[1]+c[2]+c[3]+c[4]+c[5]+c[6]+c[7]) * (1.0f/16.0f) + extra;
  float bs = block_sum(s, red);
  if (threadIdx.x == 0) partials[blockIdx.x] = bs;
}

// ---------------------------------------------------------------------------
// Kernel 2: min-projection of each 64^3 cube along one axis, pred & target.
// grid: x=16 (4096 outputs / 256), y=batch(4), z=item(24)
// ---------------------------------------------------------------------------
__global__ __launch_bounds__(256) void k_proj(const float* __restrict__ pred,
                                              const float* __restrict__ tgt,
                                              float* __restrict__ pp,
                                              float* __restrict__ pt) {
  int it = blockIdx.z;                                   // 0..23
  int b  = blockIdx.y;                                   // 0..3
  int o  = blockIdx.x * blockDim.x + threadIdx.x;        // 0..4095
  int u = o >> 6, v = o & 63;
  int cube = it / 3, ax = it % 3;
  int z0 = (cube >> 2) * 64, y0 = ((cube >> 1) & 1) * 64, x0 = (cube & 1) * 64;
  long long idx0, stp;
  if (ax == 0)      { idx0 = (((long long)b*128 +  z0    )*128 + (y0+u))*128 + (x0+v); stp = 128LL*128; }
  else if (ax == 1) { idx0 = (((long long)b*128 + (z0+u))*128 +  y0    )*128 + (x0+v); stp = 128; }
  else              { idx0 = (((long long)b*128 + (z0+u))*128 + (y0+v))*128 +  x0;     stp = 1; }
  float mp = 3.4e38f, mt = 3.4e38f;
  for (int d = 0; d < 64; ++d) {
    mp = fminf(mp, pred[idx0 + d*stp]);
    mt = fminf(mt, tgt [idx0 + d*stp]);
  }
  long long oidx = ((long long)it * 4 + b) * 4096 + o;
  pp[oidx] = mp; pt[oidx] = mt;
}

// ---------------------------------------------------------------------------
// Kernel 3: MALIS window weights + conn contribution. One block per
// (item, batch, 32x32 window). Entire state lives in LDS (~56 KB).
// Window tiles arrive via the Tensor Data Mover (async DMA to LDS).
// ---------------------------------------------------------------------------
__device__ __forceinline__ void edge_nodes(int e, int& a, int& b) {
  if (e < 992) { int r = e / 31, c = e - r*31; a = r*32 + c; b = a + 1; }
  else { int e2 = e - 992; int r = e2 >> 5, c = e2 & 31; a = r*32 + c; b = a + 32; }
}

__global__ __launch_bounds__(256) void k_malis(const float* __restrict__ pp,
                                               const float* __restrict__ pt,
                                               float* __restrict__ connp) {
  __shared__ __align__(16) float pw[1024];
  __shared__ __align__(16) float tw[1024];
  __shared__ unsigned char  m0[1024], m1[1024];
  __shared__ unsigned short seg[1024];
  __shared__ float cbase[1984], gte[1984], wgt[1984];
  __shared__ float ckey[2048];
  __shared__ unsigned short cidx[2048];
  __shared__ unsigned short par[1024], nxt[1024], tailp[1024], tot[1024];
  __shared__ float red[8];
  __shared__ int   changed;
  __shared__ float snorm;

  const int tid = threadIdx.x;
  const int it = blockIdx.z, b = blockIdx.y;
  const int wk = blockIdx.x >> 1, wj = blockIdx.x & 1;
  const long long pbase = ((long long)it * 4 + b) * 4096;
  const long long tile0 = pbase + (long long)(wk*32) * 64 + wj*32;

#if __has_builtin(__builtin_amdgcn_tensor_load_to_lds) && \
    __has_builtin(__builtin_amdgcn_s_wait_tensorcnt)
  // --- TDM: DMA both 32x32 tiles (stride-64 tensor) straight into LDS ---
  if (tid < 32) {                       // single wave issues (EXEC ignored)
    unsigned pw_lds = (unsigned)(unsigned long long)(void*)pw;  // LDS ap. low32
    unsigned tw_lds = (unsigned)(unsigned long long)(void*)tw;
    tdm_load_tile_32x32_f32(pw_lds, (unsigned long long)(pp + tile0));
    tdm_load_tile_32x32_f32(tw_lds, (unsigned long long)(pt + tile0));
    __builtin_amdgcn_s_wait_tensorcnt(0);
  }
  __syncthreads();
  for (int n = tid; n < 1024; n += 256)
    m0[n] = (tw[n] == 0.0f) ? 1 : 0;
#else
  // --- fallback: plain coalesced loads ---
  for (int n = tid; n < 1024; n += 256) {
    int r = n >> 5, c = n & 31;
    long long gi = pbase + (long long)(wk*32 + r) * 64 + (wj*32 + c);
    pw[n] = pp[gi];
    tw[n] = pt[gi];
    m0[n] = (tw[n] == 0.0f) ? 1 : 0;
  }
#endif
  __syncthreads();

  // --- dilate (t==0) 5x with 4-neighborhood; result lands in m1 ---
  for (int itr = 0; itr < 5; ++itr) {
    unsigned char* src = (itr & 1) ? m1 : m0;
    unsigned char* dst = (itr & 1) ? m0 : m1;
    for (int n = tid; n < 1024; n += 256) {
      int r = n >> 5, c = n & 31;
      unsigned char v = src[n];
      if (r > 0)  v |= src[n-32];
      if (r < 31) v |= src[n+32];
      if (c > 0)  v |= src[n-1];
      if (c < 31) v |= src[n+1];
      dst[n] = v;
    }
    __syncthreads();
  }
  for (int n = tid; n < 1024; n += 256) m0[n] = m1[n] ? 0 : 1;  // foreground
  __syncthreads();

  // --- 8-connected components: min-index label propagation to fixpoint ---
  for (int n = tid; n < 1024; n += 256) seg[n] = m0[n] ? (unsigned short)(n + 1) : 0;
  __syncthreads();
  for (int itr = 0; itr < 2048; ++itr) {
    if (tid == 0) changed = 0;
    __syncthreads();
    int loc = 0;
    for (int n = tid; n < 1024; n += 256) {
      if (!m0[n]) continue;
      int r = n >> 5, c = n & 31;
      unsigned short mn = seg[n];
      for (int dr = -1; dr <= 1; ++dr)
        for (int dc = -1; dc <= 1; ++dc) {
          int rr = r + dr, cc = c + dc;
          if (rr < 0 || rr > 31 || cc < 0 || cc > 31) continue;
          int nb = rr*32 + cc;
          if (m0[nb] && seg[nb] < mn) mn = seg[nb];
        }
      if (mn < seg[n]) { seg[n] = mn; loc = 1; }
    }
    if (loc) changed = 1;
    __syncthreads();
    int ch = changed;
    __syncthreads();
    if (!ch) break;
  }

  // --- edge costs and (uncapped) edge ground truth ---
  for (int e = tid; e < 1984; e += 256) {
    int a, bb; edge_nodes(e, a, bb);
    cbase[e] = pw[a] + pw[bb];
    gte[e]   = tw[a] + tw[bb];
  }
  __syncthreads();

  v16h av;
  #pragma unroll
  for (int i = 0; i < 16; ++i) av[i] = (_Float16)0.0f;

  for (int run = 0; run < 2; ++run) {          // run 0 = neg, run 1 = pos
    // keys (apply NCAP/PCAP cost edits), pad to 2048 with -inf
    for (int e = tid; e < 2048; e += 256) {
      if (e < 1984) {
        float g = gte[e], cst = cbase[e];
        if (run == 0) { if (g > 10.0f) cst = 10.0f; }
        else          { if (g <  3.0f) cst = 0.0f;  }
        ckey[e] = cst;
      } else ckey[e] = -3.4e38f;
      cidx[e] = (unsigned short)e;
    }
    __syncthreads();

    // bitonic sort: descending key, ascending index on ties (== stable argsort)
    for (unsigned k = 2; k <= 2048; k <<= 1) {
      for (unsigned j = k >> 1; j > 0; j >>= 1) {
        for (unsigned i = (unsigned)tid; i < 2048; i += 256) {
          unsigned x = i ^ j;
          if (x > i) {
            float ka = ckey[i], kb = ckey[x];
            unsigned short ia = cidx[i], ib = cidx[x];
            bool aFirst = (ka > kb) || (ka == kb && ia < ib);
            bool up = ((i & k) == 0);
            if (up ? !aFirst : aFirst) {
              ckey[i] = kb; ckey[x] = ka; cidx[i] = ib; cidx[x] = ia;
            }
          }
        }
        __syncthreads();
      }
    }

    // union-find init (member lists rooted at component root)
    for (int n = tid; n < 1024; n += 256) {
      par[n] = (unsigned short)n; nxt[n] = 0xFFFFu;
      tailp[n] = (unsigned short)n; tot[n] = m0[n] ? 1 : 0;
    }
    for (int e = tid; e < 1984; e += 256) wgt[e] = 0.f;
    __syncthreads();

    // serial Kruskal-order MALIS (lane 0); exact pair counting via lists
    if (tid == 0) {
      for (int s = 0; s < 2048; ++s) {
        int e = cidx[s];
        if (e >= 1984) continue;
        int a, bb; edge_nodes(e, a, bb);
        int ra = a;  while (par[ra] != ra) { par[ra] = par[par[ra]]; ra = par[ra]; }
        int rb = bb; while (par[rb] != rb) { par[rb] = par[par[rb]]; rb = par[rb]; }
        if (ra == rb) continue;
        int same = 0;
        for (int u = ra; u != 0xFFFF; u = nxt[u]) {
          int su = seg[u];
          if (!su) continue;
          for (int v = rb; v != 0xFFFF; v = nxt[v]) if (seg[v] == su) ++same;
        }
        int ta = tot[ra], tb = tot[rb];
        wgt[e] = (run == 1) ? (float)same : (float)(ta * tb - same);
        nxt[tailp[rb]] = (unsigned short)ra;      // splice A's list onto B
        tailp[rb] = tailp[ra];
        tot[rb] = (unsigned short)(ta + tb);
        par[ra] = (unsigned short)rb;
      }
    }
    __syncthreads();

    // normalize then mask (order matters: sum includes pre-mask weights)
    float ls = 0.f;
    for (int e = tid; e < 1984; e += 256) ls += wgt[e];
    float sum = block_sum(ls, red);
    if (tid == 0) snorm = (sum > 0.f) ? (1.0f / sum) : 1.0f;
    __syncthreads();
    float inv = snorm;
    for (int e = tid; e < 1984; e += 256) {
      float g = gte[e];
      bool kill = (run == 0) ? (g >= 3.0f) : (g < 10.0f);
      wgt[e] = kill ? 0.f : wgt[e] * inv;
    }
    __syncthreads();

    // scatter edge weights to nodes; pack contributions for WMMA reduce
    for (int q = 0; q < 4; ++q) {
      int n = q * 256 + tid;
      int r = n >> 5, c = n & 31;
      float acc = 0.f;
      if (c > 0)  acc += wgt[r*31 + (c-1)];
      if (c < 31) acc += wgt[r*31 + c];
      if (r > 0)  acc += wgt[992 + (r-1)*32 + c];
      if (r < 31) acc += wgt[992 + r*32 + c];
      float pv = pw[n];
      float contrib = (run == 0) ? (pv * pv * acc)
                                 : (BETA_C * (DMAX_C - pv) * (DMAX_C - pv) * acc);
      av[run*4 + q] = (_Float16)contrib;
    }
    __syncthreads();
  }

  // WMMA ones-contraction reduction of the packed contributions
  v16h ones;
  #pragma unroll
  for (int i = 0; i < 16; ++i) ones[i] = (_Float16)1.0f;
  v8f cc = {};
  cc = __builtin_amdgcn_wmma_f32_16x16x32_f16(false, av, false, ones,
                                              (short)0, cc, false, false);
  float s = cc[0]+cc[1]+cc[2]+cc[3]+cc[4]+cc[5]+cc[6]+cc[7];
  float bs = block_sum(s, red);
  if (tid == 0) {
    int task = (blockIdx.z * 4 + blockIdx.y) * 4 + blockIdx.x;
    connp[task] = bs * (1.0f / 16.0f);
  }
}

// ---------------------------------------------------------------------------
// Kernel 4: finalize -> d_out = { mse, ALPHA*conn }
// ---------------------------------------------------------------------------
__global__ __launch_bounds__(256) void k_finalize(const float* __restrict__ msep,
                                                  const float* __restrict__ connp,
                                                  float* __restrict__ out,
                                                  float invN) {
  __shared__ float sd[256];
  int tid = threadIdx.x;
  float a = 0.f;
  for (int i = tid; i < NB_MSE; i += 256) a += msep[i];
  sd[tid] = a; __syncthreads();
  for (int s = 128; s > 0; s >>= 1) { if (tid < s) sd[tid] += sd[tid + s]; __syncthreads(); }
  float mse = sd[0];
  __syncthreads();
  float b = 0.f;
  for (int i = tid; i < NTASK; i += 256) b += connp[i];
  sd[tid] = b; __syncthreads();
  for (int s = 128; s > 0; s >>= 1) { if (tid < s) sd[tid] += sd[tid + s]; __syncthreads(); }
  if (tid == 0) { out[0] = mse * invN; out[1] = ALPHA_C * sd[0]; }
}

// ---------------------------------------------------------------------------
extern "C" void kernel_launch(void* const* d_in, const int* in_sizes, int n_in,
                              void* d_out, int out_size, void* d_ws, size_t ws_size,
                              hipStream_t stream) {
  (void)n_in; (void)out_size; (void)ws_size;
  const float* pred = (const float*)d_in[0];
  const float* tgt  = (const float*)d_in[1];
  float* out = (float*)d_out;
  float* ws  = (float*)d_ws;

  long long N = (long long)in_sizes[0];      // 4 * 1 * 128^3 = 8,388,608

  float* msep  = ws;                         // NB_MSE floats
  float* projp = ws + NB_MSE;                // 24*4*4096 floats
  float* projt = projp + 24 * 4 * 4096;      // 24*4*4096 floats
  float* connp = projt + 24 * 4 * 4096;      // NTASK floats

  k_mse     <<<NB_MSE, 256, 0, stream>>>(pred, tgt, msep, N);
  k_proj    <<<dim3(16, 4, 24), 256, 0, stream>>>(pred, tgt, projp, projt);
  k_malis   <<<dim3(4, 4, 24), 256, 0, stream>>>(projp, projt, connp);
  k_finalize<<<1, 256, 0, stream>>>(msep, connp, out, 1.0f / (float)N);
}
